// MultiheadAttention_23708219474098
// MI455X (gfx1250) — matrix operand order repaired and checked by reading, not compiled, not run on hardware
//
#include <hip/hip_runtime.h>
#include <hip/hip_bf16.h>
#include <stdint.h>

#define NE   8
#define DIN  1024
#define DHID 512
#define NT   8192         // tokens
#define MT   64           // token tile per block
#define BROW 40           // padded LDS row stride for B slabs (elems)
#define HROW 520          // padded LDS row stride for H (elems)

typedef __attribute__((ext_vector_type(16))) __bf16 v16bf;
typedef __attribute__((ext_vector_type(8)))  float  v8f;

union Frag { uint4 u[2]; v16bf v; };

#if defined(__has_builtin)
#if __has_builtin(__builtin_amdgcn_global_load_async_to_lds_b128) && \
    __has_builtin(__builtin_amdgcn_s_wait_asynccnt)
#define ASYNC_LDS 1
#endif
#endif

#ifdef ASYNC_LDS
typedef int v4i_ __attribute__((vector_size(16)));
typedef __attribute__((address_space(1))) v4i_* gv4i_p;   // global 16B vector ptr
typedef __attribute__((address_space(3))) v4i_* lv4i_p;   // LDS    16B vector ptr
#endif

__device__ __forceinline__ void cp16_lds(void* lds, const void* g) {
#ifdef ASYNC_LDS
    __builtin_amdgcn_global_load_async_to_lds_b128(
        (gv4i_p)(uintptr_t)g,
        (lv4i_p)(uint32_t)(uintptr_t)lds,
        0, 0);
#else
    *(uint4*)lds = *(const uint4*)g;
#endif
}

__device__ __forceinline__ void slab_join() {
#ifdef ASYNC_LDS
    __builtin_amdgcn_s_wait_asynccnt(0);
#endif
    __syncthreads();
}

__device__ __forceinline__ uint16_t f32_to_bf16(float f) {
    uint32_t u = __builtin_bit_cast(uint32_t, f);
    uint32_t r = u + 0x7FFFu + ((u >> 16) & 1u);   // round-to-nearest-even
    if ((u & 0x7F800000u) == 0x7F800000u) r = u;   // inf/nan: truncate
    return (uint16_t)(r >> 16);
}

// ---------------- zero output + counters ----------------
__global__ void zero_kernel(float* __restrict__ out, int n4, int* __restrict__ counts) {
    int i = blockIdx.x * blockDim.x + threadIdx.x;
    if (i < NE) counts[i] = 0;
    float4 z = make_float4(0.f, 0.f, 0.f, 0.f);
    for (int j = i; j < n4; j += gridDim.x * blockDim.x)
        ((float4*)out)[j] = z;
}

// ---------------- f32 -> bf16 (x, row major) ----------------
__global__ void cvt_x_kernel(const float* __restrict__ x, uint16_t* __restrict__ xb, int n8) {
    int stride = gridDim.x * blockDim.x;
    for (int j = blockIdx.x * blockDim.x + threadIdx.x; j < n8; j += stride) {
        const float4* p = (const float4*)x + 2 * (size_t)j;
        float4 a = p[0], b = p[1];
        union { uint16_t h[8]; uint4 u; } o;
        o.h[0] = f32_to_bf16(a.x); o.h[1] = f32_to_bf16(a.y);
        o.h[2] = f32_to_bf16(a.z); o.h[3] = f32_to_bf16(a.w);
        o.h[4] = f32_to_bf16(b.x); o.h[5] = f32_to_bf16(b.y);
        o.h[6] = f32_to_bf16(b.z); o.h[7] = f32_to_bf16(b.w);
        ((uint4*)xb)[j] = o.u;
    }
}

// ---------------- w1 [E][IN][HID] -> bf16 transposed [E][HID][IN] ----------------
__global__ void cvt_w1_kernel(const float* __restrict__ w, uint16_t* __restrict__ wt) {
    const size_t total = (size_t)NE * DHID * DIN;
    size_t stride = (size_t)gridDim.x * blockDim.x;
    for (size_t idx = (size_t)blockIdx.x * blockDim.x + threadIdx.x; idx < total; idx += stride) {
        int k = (int)(idx % DIN);
        size_t r = idx / DIN;
        int n = (int)(r % DHID);
        int e = (int)(r / DHID);
        wt[idx] = f32_to_bf16(w[((size_t)e * DIN + k) * DHID + n]);
    }
}

// ---------------- w2 [E][HID][IN] -> bf16 transposed [E][IN][HID] ----------------
__global__ void cvt_w2_kernel(const float* __restrict__ w, uint16_t* __restrict__ wt) {
    const size_t total = (size_t)NE * DIN * DHID;
    size_t stride = (size_t)gridDim.x * blockDim.x;
    for (size_t idx = (size_t)blockIdx.x * blockDim.x + threadIdx.x; idx < total; idx += stride) {
        int k = (int)(idx % DHID);
        size_t r = idx / DHID;
        int n = (int)(r % DIN);
        int e = (int)(r / DIN);
        wt[idx] = f32_to_bf16(w[((size_t)e * DHID + k) * DIN + n]);
    }
}

// ---------------- gating: softmax(QK/32) -> top2 -> compacted expert lists ----------------
__global__ __launch_bounds__(256)
void gate_kernel(const float* __restrict__ Q, const float* __restrict__ K,
                 int* __restrict__ tok_list, float* __restrict__ gate_list,
                 int* __restrict__ counts) {
    const int t = blockIdx.x;
    const int lane = threadIdx.x & 31;
    const int w = threadIdx.x >> 5;      // expert id, one wave per expert
    __shared__ float sc[NE];

    const float* q  = Q + (size_t)t * DIN;
    const float* kv = K + ((size_t)t * NE + w) * DIN;
    float s = 0.f;
    for (int i = lane; i < DIN; i += 32) s = fmaf(q[i], kv[i], s);
    for (int off = 16; off; off >>= 1) s += __shfl_xor(s, off, 32);
    if (lane == 0) sc[w] = s * (1.0f / 32.0f);   // 1/sqrt(1024)
    __syncthreads();

    if (threadIdx.x == 0) {
        float p[NE];
        float m = -1e30f;
        for (int e = 0; e < NE; ++e) {
            float v = sc[e];
            if (isnan(v)) v = 0.f;
            if (isinf(v)) v = (v > 0.f) ? 20.f : -20.f;
            p[e] = v;
            m = fmaxf(m, v);
        }
        float sum = 0.f;
        for (int e = 0; e < NE; ++e) { p[e] = __expf(p[e] - m); sum += p[e]; }
        float inv = 1.0f / sum;
        int i0 = 0; float b0 = -1.f;
        for (int e = 0; e < NE; ++e) { float v = p[e] * inv; p[e] = v; if (v > b0) { b0 = v; i0 = e; } }
        int i1 = 0; float b1 = -1.f;
        for (int e = 0; e < NE; ++e) { if (e != i0 && p[e] > b1) { b1 = p[e]; i1 = e; } }
        float denom = b0 + b1 + 1e-6f;
        float g0 = b0 / denom, g1 = b1 / denom;

        int s0 = atomicAdd(&counts[i0], 1);
        tok_list[(size_t)i0 * NT + s0]  = t;
        gate_list[(size_t)i0 * NT + s0] = g0;
        int s1 = atomicAdd(&counts[i1], 1);
        tok_list[(size_t)i1 * NT + s1]  = t;
        gate_list[(size_t)i1 * NT + s1] = g1;
    }
}

// ---------------- grouped expert FFN: out += gate * (gelu(x@w1) @ w2) ----------------
__global__ __launch_bounds__(512)
void moe_kernel(const uint16_t* __restrict__ xb,
                const uint16_t* __restrict__ w1t,   // [E][HID][IN]  (n-major, k contiguous)
                const uint16_t* __restrict__ w2t,   // [E][IN][HID]
                const int* __restrict__ tok_list,
                const float* __restrict__ gate_list,
                const int* __restrict__ counts,
                float* __restrict__ out) {
    __shared__ uint16_t Hs[MT * HROW];           // 65 KB bf16 activations (padded rows)
    __shared__ uint16_t Bs[2][512 * BROW];       // 2 x 40 KB double-buffered B k-slab

    const int e = blockIdx.y;
    const int cnt = counts[e];
    const int tileBase = blockIdx.x * MT;
    if (tileBase >= cnt) return;

    const int tid  = threadIdx.x;
    const int lane = tid & 31;
    const int wid  = tid >> 5;       // 0..15
    const int mw   = wid >> 2;       // row-wave 0..3 -> rows 16*mw..
    const int nw   = wid & 3;        // col-wave 0..3 -> cols 128*nw..
    const int lo   = lane & 15;
    const int hi   = lane >> 4;      // 0|1 (half-wave)

    const int*   tl = tok_list  + (size_t)e * NT;
    const float* gl = gate_list + (size_t)e * NT;

    // A-fragment source row for this lane (row = 16*mw + lo within tile)
    int ra   = tileBase + 16 * mw + lo;
    int tokA = tl[min(ra, cnt - 1)];

    // C/D rows j map to row = 16*mw + j + 8*hi
    float g8[8];
    int   tok8[8];
#pragma unroll
    for (int j = 0; j < 8; ++j) {
        int r  = tileBase + 16 * mw + j + 8 * hi;
        int rc = min(r, cnt - 1);
        tok8[j] = tl[rc];
        g8[j]   = (r < cnt) ? gl[rc] : 0.0f;
    }

    const uint16_t* w1e = w1t + (size_t)e * DHID * DIN;
    const uint16_t* w2e = w2t + (size_t)e * DIN * DHID;

    // stage one 512x32 bf16 k-slab: thread tid copies row tid (64B = 4 x b128)
    auto stage = [&](int buf, const uint16_t* srcRow) {
        uint16_t* l = &Bs[buf][tid * BROW];
#pragma unroll
        for (int c2 = 0; c2 < 4; ++c2)
            cp16_lds(l + c2 * 8, srcRow + c2 * 8);
    };

    // ---- GEMM1: H(64x512) = gelu(X(64x1024) @ W1) * gate ----
    int buf = 0;
    stage(0, w1e + (size_t)tid * DIN);       // slab kb=0
    slab_join();

    v8f acc[8] = {};
    const uint16_t* xrow = xb + (size_t)tokA * DIN;
    for (int kb = 0; kb < DIN; kb += 32) {
        // prefetch next slab (last GEMM1 iter prefetches GEMM2 chunk0 slab0)
        if (kb + 32 < DIN)
            stage(buf ^ 1, w1e + (size_t)tid * DIN + kb + 32);
        else
            stage(buf ^ 1, w2e + (size_t)tid * DHID);

        Frag a;
        const uint16_t* pa = xrow + kb + hi * 8;         // lanes>=16: K base +8
        a.u[0] = *(const uint4*)(pa);                     // K 0..7   (rel)
        a.u[1] = *(const uint4*)(pa + 16);                // K 16..23 (rel)
#pragma unroll
        for (int t = 0; t < 8; ++t) {
            int col = 128 * nw + 16 * t + lo;
            const uint16_t* pb = &Bs[buf][col * BROW + hi * 8];
            Frag b;
            b.u[0] = *(const uint4*)(pb);
            b.u[1] = *(const uint4*)(pb + 16);
            acc[t] = __builtin_amdgcn_wmma_f32_16x16x32_bf16(
                false, a.v, false, b.v, (short)0, acc[t], false, false);
        }
        slab_join();
        buf ^= 1;
    }

    // gelu(exact) + gate scale -> bf16 H in LDS (overlaps prefetched GEMM2 slab0)
#pragma unroll
    for (int t = 0; t < 8; ++t) {
        int col = 128 * nw + 16 * t + lo;
#pragma unroll
        for (int j = 0; j < 8; ++j) {
            int row = 16 * mw + j + 8 * hi;
            float v = acc[t][j];
            v = 0.5f * v * (1.0f + erff(v * 0.70710678118f));
            v *= g8[j];
            Hs[row * HROW + col] = f32_to_bf16(v);
        }
    }
    __syncthreads();   // publish Hs

    // ---- GEMM2: Y(64x1024) = H(64x512) @ W2, atomically accumulated into out ----
#pragma unroll 1
    for (int c = 0; c < 2; ++c) {
        v8f acc2[8] = {};
        for (int kb = 0; kb < DHID; kb += 32) {
            if (kb + 32 < DHID)
                stage(buf ^ 1, w2e + (size_t)(c * 512 + tid) * DHID + kb + 32);
            else if (c == 0)
                stage(buf ^ 1, w2e + (size_t)(512 + tid) * DHID);

            Frag a;
            const uint16_t* pa = &Hs[(16 * mw + lo) * HROW + kb + hi * 8];
            a.u[0] = *(const uint4*)(pa);
            a.u[1] = *(const uint4*)(pa + 16);
#pragma unroll
            for (int t = 0; t < 8; ++t) {
                int col = 128 * nw + 16 * t + lo;
                const uint16_t* pb = &Bs[buf][col * BROW + hi * 8];
                Frag b;
                b.u[0] = *(const uint4*)(pb);
                b.u[1] = *(const uint4*)(pb + 16);
                acc2[t] = __builtin_amdgcn_wmma_f32_16x16x32_bf16(
                    false, a.v, false, b.v, (short)0, acc2[t], false, false);
            }
            slab_join();
            buf ^= 1;
        }
#pragma unroll
        for (int t = 0; t < 8; ++t) {
            int col = c * 512 + 128 * nw + 16 * t + lo;
#pragma unroll
            for (int j = 0; j < 8; ++j) {
                if (g8[j] != 0.0f)
                    atomicAdd(out + (size_t)tok8[j] * DIN + col, acc2[t][j]);
            }
        }
    }
}

extern "C" void kernel_launch(void* const* d_in, const int* in_sizes, int n_in,
                              void* d_out, int out_size, void* d_ws, size_t ws_size,
                              hipStream_t stream) {
    const float* x  = (const float*)d_in[0];   // [T][IN]
    const float* Q  = (const float*)d_in[1];   // [T][1][IN]
    const float* K  = (const float*)d_in[2];   // [T][E][IN]
    const float* w1 = (const float*)d_in[3];   // [E][IN][HID]
    const float* w2 = (const float*)d_in[4];   // [E][HID][IN]
    float* out = (float*)d_out;

    char* ws = (char*)d_ws;
    int*      counts    = (int*)ws;                                   // 256 B
    int*      tok_list  = (int*)(ws + 256);                           // E*T*4   = 256 KB
    float*    gate_list = (float*)(ws + 256 + (size_t)NE * NT * 4);   // 256 KB
    size_t off = 256 + 2 * (size_t)NE * NT * 4;                       // 524544 (256-aligned)
    uint16_t* xb   = (uint16_t*)(ws + off);                           // T*IN*2   = 16 MB
    off += (size_t)NT * DIN * 2;
    uint16_t* w1bt = (uint16_t*)(ws + off);                           // E*HID*IN*2 = 8 MB
    off += (size_t)NE * DHID * DIN * 2;
    uint16_t* w2bt = (uint16_t*)(ws + off);                           // E*IN*HID*2 = 8 MB

    zero_kernel<<<2048, 256, 0, stream>>>(out, NT * DIN / 4, counts);
    cvt_x_kernel<<<2048, 256, 0, stream>>>(x, xb, NT * DIN / 8);
    cvt_w1_kernel<<<4096, 256, 0, stream>>>(w1, w1bt);
    cvt_w2_kernel<<<4096, 256, 0, stream>>>(w2, w2bt);
    gate_kernel<<<NT, 256, 0, stream>>>(Q, K, tok_list, gate_list, counts);

    dim3 grid(NT / MT, NE);
    moe_kernel<<<grid, 512, 0, stream>>>(xb, w1bt, w2bt, tok_list, gate_list, counts, out);
}
